// ForegroundConsistencyLoss_61040075211414
// MI455X (gfx1250) — matrix operand order repaired and checked by reading, not compiled, not run on hardware
//
#include <hip/hip_runtime.h>
#include <hip/hip_bf16.h>

typedef __attribute__((ext_vector_type(16))) _Float16 v16h;
typedef __attribute__((ext_vector_type(8)))  float    v8f;
typedef __attribute__((ext_vector_type(2)))  float    v2f;

#define BG_NEG (-1.0e18f)
#define JT 4   // column tiles per job (A-operand reuse factor)

// ---------------------------------------------------------------------------
// Prep: per point -> padded pc4 (xyz0), fused sq/foreground, f16 normalized
// logits padded K=20 -> 32.
// ---------------------------------------------------------------------------
__global__ void fcl_prep(const float* __restrict__ pc,
                         const float* __restrict__ logits,
                         float* __restrict__ pc4,
                         float* __restrict__ sqfg,
                         _Float16* __restrict__ lnh,
                         int N, int Npad)
{
    int i = blockIdx.x * blockDim.x + threadIdx.x;
    if (i >= Npad) return;
    if (i >= N) {
        pc4[i * 4 + 0] = 0.f; pc4[i * 4 + 1] = 0.f;
        pc4[i * 4 + 2] = 0.f; pc4[i * 4 + 3] = 0.f;
        sqfg[i] = BG_NEG;
        _Float16* o = lnh + (size_t)i * 32;
        #pragma unroll
        for (int c = 0; c < 32; ++c) o[c] = (_Float16)0.f;
        return;
    }
    float x = pc[i * 3 + 0], y = pc[i * 3 + 1], z = pc[i * 3 + 2];
    pc4[i * 4 + 0] = x; pc4[i * 4 + 1] = y;
    pc4[i * 4 + 2] = z; pc4[i * 4 + 3] = 0.f;
    float sq = x * x + y * y + z * z;

    const float* lg = logits + (size_t)i * 20;
    float best = lg[0];
    int   bi   = 0;
    float ss   = lg[0] * lg[0];
    #pragma unroll
    for (int c = 1; c < 20; ++c) {
        float v = lg[c];
        ss += v * v;
        if (v > best) { best = v; bi = c; }   // first-occurrence argmax
    }
    float inv = 1.0f / fmaxf(sqrtf(ss), 1e-8f);
    _Float16* o = lnh + (size_t)i * 32;
    #pragma unroll
    for (int c = 0; c < 20; ++c) o[c] = (_Float16)(lg[c] * inv);
    #pragma unroll
    for (int c = 20; c < 32; ++c) o[c] = (_Float16)0.f;

    sqfg[i] = (bi != 0) ? sq : BG_NEG;   // background => d2 hugely negative
}

// ---------------------------------------------------------------------------
__global__ void fcl_init(double* gsum, unsigned long long* gcnt)
{
    *gsum = 0.0;
    *gcnt = 0ull;
}

__global__ void fcl_final(const double* gsum, const unsigned long long* gcnt,
                          float* out)
{
    unsigned long long c = *gcnt;
    out[0] = (c > 0ull) ? (float)(*gsum / (double)c) : 1.0f;
}

// ---------------------------------------------------------------------------
// Main: each wave owns a job = (row tile ti, JT consecutive column tiles).
// A-side operands (f16 rows, f32 points, sq values) loaded once per job.
// Per tile:  v_wmma_f32_16x16x32_f16 (sim) + v_wmma_f32_16x16x4_f32 (dot).
// Count via ballot+popcount (wave-uniform, SALU). Symmetry: tj >= ti only.
// ---------------------------------------------------------------------------
__global__ void __launch_bounds__(256)
fcl_tiles(const float* __restrict__ pc4,
          const float* __restrict__ sqfg,
          const _Float16* __restrict__ lnh,
          double* __restrict__ gsum,
          unsigned long long* __restrict__ gcnt,
          int ntile)
{
    const int lane   = threadIdx.x & 31;
    const int wid    = (blockIdx.x * blockDim.x + threadIdx.x) >> 5;
    const int nwaves = (gridDim.x * blockDim.x) >> 5;
    const int g      = lane >> 4;     // lane group
    const int lm     = lane & 15;
    const int ncj    = (ntile + JT - 1) / JT;
    const int NJOBS  = ntile * ncj;

    float              accS = 0.0f;
    unsigned long long accC = 0ull;   // wave-uniform (ballot-based)

    for (int job = wid; job < NJOBS; job += nwaves) {
        const int ti  = job / ncj;
        const int tj0 = (job - ti * ncj) * JT;
        if (tj0 + JT - 1 < ti) continue;          // whole job below diagonal
        const int i0 = ti * 16;

        // ---- A-side (invariant across the JT tiles) -----------------------
        v16h A;
        {
            const _Float16* arow = lnh + (size_t)(i0 + lm) * 32;
            #pragma unroll
            for (int e = 0; e < 8; ++e) A[e]     = arow[8 * g + e];
            #pragma unroll
            for (int e = 0; e < 8; ++e) A[8 + e] = arow[16 + 8 * g + e];
        }
        v2f pa;
        {
            const float* par = pc4 + (size_t)(i0 + lm) * 4 + 2 * g;
            pa[0] = par[0]; pa[1] = par[1];
        }
        float sqm[8];
        #pragma unroll
        for (int r = 0; r < 8; ++r) sqm[r] = sqfg[i0 + r + 8 * g];

        for (int jt = 0; jt < JT; ++jt) {
            const int tj = tj0 + jt;
            if (tj < ti || tj >= ntile) continue; // wave-uniform skip
            const int j0  = tj * 16;
            const int wgt = (ti == tj) ? 1 : 2;

            // ---- B operands for this column tile --------------------------
            v16h B;
            {
                const _Float16* brow = lnh + (size_t)(j0 + lm) * 32;
                #pragma unroll
                for (int e = 0; e < 8; ++e) B[e]     = brow[8 * g + e];
                #pragma unroll
                for (int e = 0; e < 8; ++e) B[8 + e] = brow[16 + 8 * g + e];
            }
            v2f pb;
            {
                const float* pbr = pc4 + (size_t)(j0 + lm) * 4 + 2 * g;
                pb[0] = pbr[0]; pb[1] = pbr[1];
            }
            const float sqn = sqfg[j0 + lm];

            v8f cz = {};
            v8f sim = __builtin_amdgcn_wmma_f32_16x16x32_f16(
                          false, A, false, B, (short)0, cz, false, false);
            v8f dz = {};
            v8f dotv = __builtin_amdgcn_wmma_f32_16x16x4_f32(
                           false, pa, false, pb, (short)0, dz, false, false);

            // ---- mask + accumulate: D layout (M = r + 8g, N = lm) ---------
            float              ts = 0.0f;
            unsigned long long tb = 0ull;
            #pragma unroll
            for (int r = 0; r < 8; ++r) {
                const float d2 = sqm[r] + sqn - 2.0f * dotv[r];
                const bool  m  = d2 > 4.0f;       // DIST_THRESH^2; bg auto-fails
                ts += m ? fmaxf(sim[r], 0.0f) : 0.0f;
                tb += (unsigned long long)__popcll(__ballot(m));
            }
            accS += (float)wgt * ts;
            accC += (unsigned long long)wgt * tb;
        }
    }

    // wave32 reduction of the per-lane sum (count is already wave-uniform)
    #pragma unroll
    for (int off = 16; off > 0; off >>= 1)
        accS += __shfl_xor(accS, off, 32);

    if (lane == 0) {
        atomicAdd(gsum, (double)accS);
        atomicAdd(gcnt, accC);
    }
}

// ---------------------------------------------------------------------------
extern "C" void kernel_launch(void* const* d_in, const int* in_sizes, int n_in,
                              void* d_out, int out_size, void* d_ws, size_t ws_size,
                              hipStream_t stream)
{
    (void)n_in; (void)out_size; (void)ws_size;
    const float* pc     = (const float*)d_in[0];   // [N,3]
    const float* logits = (const float*)d_in[1];   // [N,20]
    float*       out    = (float*)d_out;

    const int N     = in_sizes[0] / 3;
    const int Npad  = (N + 15) & ~15;
    const int ntile = Npad / 16;

    // workspace layout (needs Npad*84 + 16 bytes; ~688KB for N=8192)
    char* ws = (char*)d_ws;
    _Float16* lnh  = (_Float16*)(ws);                          // Npad*32 halves
    float*    pc4  = (float*)(ws + (size_t)Npad * 64);         // Npad*4 floats
    float*    sqfg = (float*)(ws + (size_t)Npad * 80);         // Npad floats
    double*   gsum = (double*)(ws + (size_t)Npad * 84);
    unsigned long long* gcnt = (unsigned long long*)(ws + (size_t)Npad * 84 + 8);

    fcl_init<<<1, 1, 0, stream>>>(gsum, gcnt);
    fcl_prep<<<(Npad + 255) / 256, 256, 0, stream>>>(pc, logits, pc4, sqfg, lnh,
                                                     N, Npad);
    // 512 blocks x 8 waves = 4096 waves grid-striding over the job space
    fcl_tiles<<<512, 256, 0, stream>>>(pc4, sqfg, lnh, gsum, gcnt, ntile);
    fcl_final<<<1, 1, 0, stream>>>(gsum, gcnt, out);
}
